// QuantRNet_64201171141237
// MI455X (gfx1250) — 8-bit matrix operand order repaired and checked by reading, compiled, not run on hardware
//
#include <hip/hip_runtime.h>
#include <cstdint>

typedef __attribute__((ext_vector_type(8))) int v8i;

constexpr int    BN = 4096;
constexpr size_t OFF_SCALES = 0;
constexpr size_t OFF_W1 = 256;
constexpr size_t OFF_W2 = OFF_W1 + 32 * 64;
constexpr size_t OFF_W3 = OFF_W2 + 48 * 576;
constexpr size_t OFF_WF = OFF_W3 + 64 * 256;
constexpr size_t OFF_WH = OFF_WF + 128 * 576;
constexpr size_t OFF_A0 = OFF_WH + 16 * 128;
constexpr size_t OFF_C1 = OFF_A0 + (size_t)BN * 24 * 24 * 3;
constexpr size_t OFF_P1 = OFF_C1 + (size_t)BN * 22 * 22 * 32;
constexpr size_t OFF_C2 = OFF_P1 + (size_t)BN * 11 * 11 * 64;
constexpr size_t OFF_P2 = OFF_C2 + (size_t)BN * 9 * 9 * 48;
constexpr size_t OFF_A3 = OFF_P2 + (size_t)BN * 4 * 4 * 64;
constexpr size_t OFF_A4 = OFF_A3 + (size_t)BN * 576;

__global__ __launch_bounds__(256)
void quant_weights(const float* w1, const float* w2, const float* w3,
                   const float* fw, const float* bw, const float* cw,
                   int8_t* W1, int8_t* W2, int8_t* W3, int8_t* WF, int8_t* WH,
                   float* scales)
{
    __shared__ float red[256];
    __shared__ float s_sh;

    const float* src = nullptr; int8_t* dst = nullptr;
    int N = 0, K = 0, Kpad = 0, KHKW = 1, mode = 2, icmul = 0;
    int8_t* zptr = nullptr; int zlen = 0;
    switch (blockIdx.x) {
        case 0: src = w1; N = 28;  K = 27;  Kpad = 64;  KHKW = 9; mode = 1; icmul = 3;
                dst = W1; zptr = W1; zlen = 32 * 64;  break;
        case 1: src = w2; N = 48;  K = 252; Kpad = 576; KHKW = 9; mode = 0;
                dst = W2; zptr = W2; zlen = 48 * 576; break;
        case 2: src = w3; N = 64;  K = 192; Kpad = 256; KHKW = 4; mode = 0;
                dst = W3; zptr = W3; zlen = 64 * 256; break;
        case 3: src = fw; N = 128; K = 576; Kpad = 576; mode = 2;
                dst = WF; zptr = WF; zlen = 128 * 576; break;
        case 4: src = bw; N = 4;   K = 128; Kpad = 128; mode = 2;
                dst = WH;           zptr = WH + 6 * 128; zlen = 10 * 128; break;
        case 5: src = cw; N = 2;   K = 128; Kpad = 128; mode = 2;
                dst = WH + 4 * 128; zptr = nullptr;      zlen = 0;        break;
        default: return;
    }
    const int tid = threadIdx.x;
    for (int i = tid; i < zlen; i += 256) zptr[i] = 0;

    const int nelem = N * K;
    float mx = 0.f;
    for (int i = tid; i < nelem; i += 256) { float a = fabsf(src[i]); mx = a > mx ? a : mx; }
    red[tid] = mx;
    __syncthreads();
    for (int st = 128; st > 0; st >>= 1) {
        if (tid < st) red[tid] = fmaxf(red[tid], red[tid + st]);
        __syncthreads();
    }
    if (tid == 0) { s_sh = red[0] / 7.0f; scales[blockIdx.x] = s_sh; }
    __syncthreads();
    const float inv = 1.0f / s_sh;
    for (int i = tid; i < nelem; i += 256) {
        int n = i / K, r = i - n * K;
        int dk;
        if (mode == 0)      { int ic = r / KHKW; int rr = r - ic * KHKW; dk = rr * 64 + ic; }
        else if (mode == 1) { int ic = r / KHKW; int rr = r - ic * KHKW; dk = rr * icmul + ic; }
        else                { dk = r; }
        int q = __float2int_rn(src[i] * inv);
        q = q < -7 ? -7 : (q > 7 ? 7 : q);
        dst[(size_t)n * Kpad + dk] = (int8_t)q;
    }
}

__global__ __launch_bounds__(256)
void quant_act_nhwc(const float* __restrict__ x, const float* __restrict__ s,
                    int8_t* __restrict__ out, int n)
{
    const float inv = 1.0f / s[0];
    for (int i = blockIdx.x * 256 + threadIdx.x; i < n; i += gridDim.x * 256) {
        int c = i % 3;  int p = i / 3;
        int w = p % 24; p /= 24;
        int h = p % 24; int b = p / 24;
        int q = __float2int_rn(x[(((size_t)b * 3 + c) * 24 + h) * 24 + w] * inv);
        q = q < -8 ? -8 : (q > 7 ? 7 : q);
        out[i] = (int8_t)q;
    }
}

template <int H, int W, int OH, int OW, int CS_IN, int C, int CPAD>
__global__ __launch_bounds__(256)
void maxpool3s2_nhwc(const int8_t* __restrict__ in, int8_t* __restrict__ out, int Bn)
{
    const int total = Bn * OH * OW * CPAD;
    for (int i = blockIdx.x * 256 + threadIdx.x; i < total; i += gridDim.x * 256) {
        int c  = i % CPAD; int p = i / CPAD;
        int ow = p % OW;   p /= OW;
        int oh = p % OH;   int b = p / OH;
        int8_t mx = 0;
        if (c < C) {
            int y0 = oh * 2, x0 = ow * 2;
            int y1 = y0 + 3 < H ? y0 + 3 : H;
            int x1 = x0 + 3 < W ? x0 + 3 : W;
            for (int y = y0; y < y1; ++y)
                for (int xx = x0; xx < x1; ++xx) {
                    int8_t v = in[(((size_t)b * H + y) * W + xx) * CS_IN + c];
                    if (v > mx) mx = v;
                }
        }
        out[i] = mx;
    }
}

template <int IC, int IH, int IW, int KH, int KW,
          int OC, int OH, int OW, int Kreal, int Kpad, int NT,
          int MODE, int OCS, int TRANS>
__global__ __launch_bounds__(32)
void conv_wmma_iu8(const int8_t* __restrict__ act, const int8_t* __restrict__ W,
                   const float* __restrict__ bias,
                   const float* __restrict__ a_in, const float* __restrict__ w_scale,
                   const float* __restrict__ a_out,
                   int8_t* __restrict__ out)
{
    const int lane = threadIdx.x;
    const int l16  = lane & 15;
    const int half = lane >> 4;
    const int tm   = blockIdx.x;

    const int m_a = tm * 16 + l16;
    const int ab  = m_a / (OH * OW);
    const int ar  = m_a - ab * (OH * OW);
    const int aoy = ar / OW;
    const int aox = ar - aoy * OW;

    v8i acc[NT];
    #pragma unroll
    for (int t = 0; t < NT; ++t) acc[t] = (v8i){};

    #pragma unroll
    for (int s = 0; s < Kpad; s += 64) {
        v8i afrag;
        if constexpr (MODE == 2) {
            const int8_t* ap = act + (size_t)m_a * IC + s + half * 8;
            #pragma unroll
            for (int j = 0; j < 4; ++j) {
                int2 t2 = *(const int2*)(ap + 16 * j);
                afrag[2 * j] = t2.x; afrag[2 * j + 1] = t2.y;
            }
        } else if constexpr (MODE == 1) {
            const int g  = s / 64;
            const int ky = g / KW, kx = g - ky * KW;
            const int8_t* ap = act
                + (((size_t)ab * IH + (aoy + ky)) * IW + (aox + kx)) * IC
                + half * 8;
            #pragma unroll
            for (int j = 0; j < 4; ++j) {
                int2 t2 = *(const int2*)(ap + 16 * j);
                afrag[2 * j] = t2.x; afrag[2 * j + 1] = t2.y;
            }
        } else {
            constexpr int ROWB = KW * IC;
            const size_t abase = ((size_t)ab * IH + aoy) * IW + aox;
            #pragma unroll
            for (int j = 0; j < 4; ++j) {
                uint32_t lo = 0, hi = 0;
                #pragma unroll
                for (int t = 0; t < 8; ++t) {
                    const int k = s + half * 8 + 16 * j + t;
                    uint32_t by = 0;
                    if (k < Kreal) {
                        const int ky = k / ROWB;
                        const int r  = k - ky * ROWB;
                        by = (uint8_t)act[(abase + (size_t)ky * IW) * IC + r];
                    }
                    if (t < 4) lo |= by << (t * 8);
                    else       hi |= by << ((t - 4) * 8);
                }
                afrag[2 * j] = (int)lo; afrag[2 * j + 1] = (int)hi;
            }
        }
        #pragma unroll
        for (int t = 0; t < NT; ++t) {
            const int8_t* wp = W + (size_t)(t * 16 + l16) * Kpad + half * 8 + s;
            if (s + 64 < Kpad) __builtin_prefetch(wp + 64, 0, 1);
            v8i bfrag;
            #pragma unroll
            for (int j = 0; j < 4; ++j) {
                int2 b2 = *(const int2*)(wp + 16 * j);
                bfrag[2 * j] = b2.x; bfrag[2 * j + 1] = b2.y;
            }
            acc[t] = __builtin_amdgcn_wmma_i32_16x16x64_iu8(true, afrag, true, bfrag,
                                                            acc[t], false, false);
        }
    }

    const float sc   = a_in[0] * w_scale[0];
    const float mult = sc / a_out[0];
    size_t oidx[8];
    #pragma unroll
    for (int v = 0; v < 8; ++v) {
        const int m  = tm * 16 + v + 8 * half;
        const int b  = m / (OH * OW);
        const int r  = m - b * (OH * OW);
        const int oy = r / OW, ox = r - oy * OW;
        if (TRANS) oidx[v] = (size_t)b * OC * (OH * OW) + ox * OH + oy;
        else       oidx[v] = (((size_t)b * OH + oy) * OW + ox) * OCS;
    }
    #pragma unroll
    for (int t = 0; t < NT; ++t) {
        const int n = t * 16 + l16;
        if (n < OC) {
            const int bint = __float2int_rn(bias[n] / sc);
            #pragma unroll
            for (int v = 0; v < 8; ++v) {
                int q = __float2int_rn((float)(acc[t][v] + bint) * mult);
                q = q < 0 ? 0 : (q > 15 ? 15 : q);
                if (TRANS) out[oidx[v] + (size_t)n * (OH * OW)] = (int8_t)q;
                else       out[oidx[v] + n]                     = (int8_t)q;
            }
        }
    }
}

__global__ __launch_bounds__(32)
void heads_wmma_iu8(const int8_t* __restrict__ act4, const int8_t* __restrict__ WH,
                    const float* __restrict__ bb, const float* __restrict__ cb,
                    const float* __restrict__ s4, const float* __restrict__ scales,
                    const float* __restrict__ s_bbox, const float* __restrict__ s_conf,
                    float* __restrict__ out, int Bn)
{
    const int lane = threadIdx.x;
    const int l16  = lane & 15;
    const int half = lane >> 4;
    const int tm   = blockIdx.x;

    v8i acc = {};
    #pragma unroll
    for (int s = 0; s < 128; s += 64) {
        const int m = tm * 16 + l16;
        const int8_t* ap = act4 + (size_t)m * 128 + half * 8 + s;
        v8i afrag;
        #pragma unroll
        for (int j = 0; j < 4; ++j) {
            int2 t2 = *(const int2*)(ap + 16 * j);
            afrag[2 * j] = t2.x; afrag[2 * j + 1] = t2.y;
        }
        const int8_t* wp = WH + (size_t)l16 * 128 + half * 8 + s;
        v8i bfrag;
        #pragma unroll
        for (int j = 0; j < 4; ++j) {
            int2 b2 = *(const int2*)(wp + 16 * j);
            bfrag[2 * j] = b2.x; bfrag[2 * j + 1] = b2.y;
        }
        acc = __builtin_amdgcn_wmma_i32_16x16x64_iu8(true, afrag, true, bfrag, acc,
                                                     false, false);
    }

    const float a4 = s4[0];
    const int n = l16;
    #pragma unroll
    for (int v = 0; v < 8; ++v) {
        int m = tm * 16 + v + 8 * half;
        if (n < 4) {
            float sc = a4 * scales[4];
            int bint = __float2int_rn(bb[n] / sc);
            float x = (float)(acc[v] + bint) * sc;
            float sb = s_bbox[0];
            int q = __float2int_rn(x / sb);
            q = q < -8 ? -8 : (q > 7 ? 7 : q);
            out[(size_t)m * 4 + n] = (float)q * sb;
        } else if (n < 6) {
            float sc = a4 * scales[5];
            int bint = __float2int_rn(cb[n - 4] / sc);
            float x = (float)(acc[v] + bint) * sc;
            float sb = s_conf[0];
            int q = __float2int_rn(x / sb);
            q = q < -8 ? -8 : (q > 7 ? 7 : q);
            out[(size_t)Bn * 4 + (size_t)m * 2 + (n - 4)] = (float)q * sb;
        }
    }
}

extern "C" void kernel_launch(void* const* d_in, const int* in_sizes, int n_in,
                              void* d_out, int out_size, void* d_ws, size_t ws_size,
                              hipStream_t stream)
{
    (void)in_sizes; (void)n_in; (void)out_size; (void)ws_size;
    const float* x    = (const float*)d_in[0];
    const float* w1   = (const float*)d_in[1];
    const float* b1   = (const float*)d_in[2];
    const float* w2   = (const float*)d_in[3];
    const float* b2   = (const float*)d_in[4];
    const float* w3   = (const float*)d_in[5];
    const float* b3   = (const float*)d_in[6];
    const float* fw1  = (const float*)d_in[7];
    const float* fb1  = (const float*)d_in[8];
    const float* bw   = (const float*)d_in[9];
    const float* bb   = (const float*)d_in[10];
    const float* cw   = (const float*)d_in[11];
    const float* cb   = (const float*)d_in[12];
    const float* s_inp  = (const float*)d_in[13];
    const float* s1     = (const float*)d_in[14];
    const float* s2     = (const float*)d_in[15];
    const float* s3     = (const float*)d_in[16];
    const float* s4     = (const float*)d_in[17];
    const float* s_bbox = (const float*)d_in[18];
    const float* s_conf = (const float*)d_in[19];

    uint8_t* ws = (uint8_t*)d_ws;
    float*  scales = (float*)(ws + OFF_SCALES);
    int8_t* W1 = (int8_t*)(ws + OFF_W1);
    int8_t* W2 = (int8_t*)(ws + OFF_W2);
    int8_t* W3 = (int8_t*)(ws + OFF_W3);
    int8_t* WF = (int8_t*)(ws + OFF_WF);
    int8_t* WH = (int8_t*)(ws + OFF_WH);
    int8_t* A0 = (int8_t*)(ws + OFF_A0);
    int8_t* C1 = (int8_t*)(ws + OFF_C1);
    int8_t* P1 = (int8_t*)(ws + OFF_P1);
    int8_t* C2 = (int8_t*)(ws + OFF_C2);
    int8_t* P2 = (int8_t*)(ws + OFF_P2);
    int8_t* A3 = (int8_t*)(ws + OFF_A3);
    int8_t* A4 = (int8_t*)(ws + OFF_A4);

    quant_weights<<<6, 256, 0, stream>>>(w1, w2, w3, fw1, bw, cw,
                                         W1, W2, W3, WF, WH, scales);
    {
        int n = BN * 24 * 24 * 3;
        quant_act_nhwc<<<(n + 255) / 256, 256, 0, stream>>>(x, s_inp, A0, n);
    }
    conv_wmma_iu8<3, 24, 24, 3, 3, 28, 22, 22, 27, 64, 2, 0, 32, 0>
        <<<BN * 22 * 22 / 16, 32, 0, stream>>>(A0, W1, b1, s_inp, scales + 0, s1, C1);
    {
        int total = BN * 11 * 11 * 64;
        maxpool3s2_nhwc<22, 22, 11, 11, 32, 28, 64>
            <<<(total + 255) / 256, 256, 0, stream>>>(C1, P1, BN);
    }
    conv_wmma_iu8<64, 11, 11, 3, 3, 48, 9, 9, 576, 576, 3, 1, 48, 0>
        <<<BN * 9 * 9 / 16, 32, 0, stream>>>(P1, W2, b2, s1, scales + 1, s2, C2);
    {
        int total = BN * 4 * 4 * 64;
        maxpool3s2_nhwc<9, 9, 4, 4, 48, 48, 64>
            <<<(total + 255) / 256, 256, 0, stream>>>(C2, P2, BN);
    }
    conv_wmma_iu8<64, 4, 4, 2, 2, 64, 3, 3, 256, 256, 4, 1, 0, 1>
        <<<BN * 3 * 3 / 16, 32, 0, stream>>>(P2, W3, b3, s2, scales + 2, s3, A3);
    conv_wmma_iu8<576, 1, 1, 1, 1, 128, 1, 1, 576, 576, 8, 2, 128, 0>
        <<<BN / 16, 32, 0, stream>>>(A3, WF, fb1, s3, scales + 3, s4, A4);
    heads_wmma_iu8<<<BN / 16, 32, 0, stream>>>(A4, WH, bb, cb, s4, scales,
                                               s_bbox, s_conf, (float*)d_out, BN);
}
